// EBP_binaryNetRelaxed_29283087024619
// MI455X (gfx1250) — compile-verified, matrix operand order checked by reading, not run on hardware
//
#include <hip/hip_runtime.h>
#include <hip/hip_bf16.h>

typedef __attribute__((ext_vector_type(16))) _Float16 v16h;
typedef __attribute__((ext_vector_type(8)))  float    v8f;
typedef __attribute__((ext_vector_type(4)))  float    f32x4;   // native vector for NT stores

#define H_DIM   1024
#define M_DIM   128
#define D_OUT   10
#define D_IN    784
#define K0_PAD  800            // 784 padded to multiple of 32
#define BETA    0.2f
#define PION2   1.570796326794f
#define CCOEF   (PION2 * BETA * BETA)

// ---------------------------------------------------------------------------
// Fragment loader for wave32 V_WMMA_F32_16X16X32_F16 (A 16x32 / B 32x16).
// A row (or B column, stored K-contiguous) layout per lane:
//   lane&15 = M (or N); kgroup = (lane>>4)*8
//   VGPR i<4  : K = kgroup + 2i, 2i+1
//   VGPR i>=4 : K = 16 + kgroup + 2(i-4), +1
// Each VGPR is one aligned 32-bit load of a contiguous f16 pair.
// ---------------------------------------------------------------------------
union FragU { v16h v; unsigned u[8]; };

__device__ __forceinline__ v16h load_frag16(const _Float16* __restrict__ rowbase,
                                            int k0, int lane) {
  const int kg = (lane >> 4) << 3;              // 0 or 8
  const unsigned* p = (const unsigned*)(rowbase + k0 + kg);
  FragU f;
#pragma unroll
  for (int i = 0; i < 4; ++i) f.u[i] = p[i];        // K = kg+2i
#pragma unroll
  for (int i = 4; i < 8; ++i) f.u[i] = p[i + 4];    // K = 16+kg+2(i-4)
  return f.v;
}

// ---------------------------------------------------------------------------
// x (128x784 f32) -> f16, K padded to 800 with zeros
// ---------------------------------------------------------------------------
__global__ void prep_x_kernel(const float* __restrict__ x, _Float16* __restrict__ x16) {
  int t = blockIdx.x * blockDim.x + threadIdx.x;
  if (t >= M_DIM * K0_PAD) return;
  int m = t / K0_PAD, k = t % K0_PAD;
  float v = (k < D_IN) ? x[m * D_IN + k] : 0.0f;
  x16[t] = (_Float16)v;
}

// ---------------------------------------------------------------------------
// w (din x dout f32) -> mT (dout x ldK f16, K-contiguous, zero padded),
// msqT (optional), colsum[c] = sum_r (1 - m^2)
// ---------------------------------------------------------------------------
__global__ void prep_w_kernel(const float* __restrict__ w,
                              _Float16* __restrict__ mT,
                              _Float16* __restrict__ msqT,
                              float* __restrict__ colsum,
                              int din, int dout, int ldK) {
  int c = blockIdx.x * blockDim.x + threadIdx.x;
  if (c >= dout) return;
  float s = 0.0f;
  for (int r = 0; r < ldK; ++r) {
    float mm = 0.0f;
    if (r < din) {
      float wv = w[(size_t)r * dout + c];
      float sg = 1.0f / (1.0f + __expf(-wv));
      mm = 2.0f * sg - 1.0f;
      s += 1.0f - mm * mm;
    }
    mT[(size_t)c * ldK + r] = (_Float16)mm;
    if (msqT) msqT[(size_t)c * ldK + r] = (_Float16)(mm * mm);
  }
  colsum[c] = s;
}

// ---------------------------------------------------------------------------
// Fused EBP layer: one wave per 16x16 output tile.
//   hbar    = A @ M            (WMMA f16->f32)
//   diagsig = V @ Msq + colsum (WMMA f16->f32, optional V)
//   x = tanh(beta*hbar / sqrt(1 + c*diagsig)), v = 1 - x^2
// ---------------------------------------------------------------------------
__global__ void ebp_layer_wmma(const _Float16* __restrict__ A,
                               const _Float16* __restrict__ V,      // nullable
                               const _Float16* __restrict__ MT,
                               const _Float16* __restrict__ MsqT,   // nullable
                               const float*    __restrict__ colsum,
                               const float*    __restrict__ th,
                               _Float16* __restrict__ Xout,
                               _Float16* __restrict__ Vout,
                               float*    __restrict__ Xout32,       // nullable
                               float*    __restrict__ Vout32,       // nullable
                               int ldK, int N) {
  const int lane  = threadIdx.x & 31;
  const int wave  = blockIdx.x * (blockDim.x >> 5) + (threadIdx.x >> 5);
  const int tileM = wave & 7;          // 128/16 = 8 row tiles
  const int tileN = wave >> 3;

  const int rc = lane & 15;            // A row / B col within tile
  const _Float16* arow  = A  + (size_t)(tileM * 16 + rc) * ldK;
  const _Float16* brow  = MT + (size_t)(tileN * 16 + rc) * ldK;
  const _Float16* vrow  = V    ? V    + (size_t)(tileM * 16 + rc) * ldK : nullptr;
  const _Float16* brow2 = MsqT ? MsqT + (size_t)(tileN * 16 + rc) * ldK : nullptr;

  v8f acc_h = {};
  v8f acc_s = {};
  for (int k0 = 0; k0 < ldK; k0 += 32) {
    v16h a = load_frag16(arow, k0, lane);
    v16h b = load_frag16(brow, k0, lane);
    acc_h = __builtin_amdgcn_wmma_f32_16x16x32_f16(false, a, false, b,
                                                   (short)0, acc_h, false, false);
    if (V) {
      v16h av = load_frag16(vrow,  k0, lane);
      v16h bs = load_frag16(brow2, k0, lane);
      acc_s = __builtin_amdgcn_wmma_f32_16x16x32_f16(false, av, false, bs,
                                                     (short)0, acc_s, false, false);
    }
  }

  // C/D layout: element i of v8f -> row = tileM*16 + (lane>>4)*8 + i, col = tileN*16 + (lane&15)
  const int baseRow = tileM * 16 + ((lane >> 4) << 3);
  const int col     = tileN * 16 + (lane & 15);
  const float thc   = th[col];
  const float csc   = colsum[col];
#pragma unroll
  for (int i = 0; i < 8; ++i) {
    int r = baseRow + i;
    float hbar = acc_h[i] + thc;
    float ds   = csc + (V ? acc_s[i] : 0.0f);
    float h    = BETA * hbar * rsqrtf(1.0f + CCOEF * ds);
    float xx   = tanhf(h);
    float vv   = 1.0f - xx * xx;
    size_t idx = (size_t)r * N + col;
    Xout[idx] = (_Float16)xx;
    Vout[idx] = (_Float16)vv;
    if (Xout32) { Xout32[idx] = xx; Vout32[idx] = vv; }
  }
}

// ---------------------------------------------------------------------------
// Zero the two scalar output slots (loss, frac_correct)
// ---------------------------------------------------------------------------
__global__ void zero2_kernel(float* a, float* b) {
  if (threadIdx.x == 0 && blockIdx.x == 0) { *a = 0.0f; *b = 0.0f; }
}

// ---------------------------------------------------------------------------
// Last layer (1024 -> 10), log-softmax, loss, frac_correct.
// One wave per sample; lane j < 10 computes one output column in f32.
// ---------------------------------------------------------------------------
__global__ void last_layer_kernel(const float* __restrict__ x4,
                                  const float* __restrict__ v4,
                                  const float* __restrict__ wlast,
                                  const float* __restrict__ thlast,
                                  const float* __restrict__ noise,
                                  const int*   __restrict__ target,
                                  float* __restrict__ out_h,
                                  float* __restrict__ out_lp,
                                  float* __restrict__ out_loss,
                                  float* __restrict__ out_frac) {
  __shared__ float sh[D_OUT];
  const int m = blockIdx.x;
  const int j = threadIdx.x;
  float h = 0.0f;
  if (j < D_OUT) {
    float hb = 0.0f, ds = 0.0f, cs = 0.0f;
    for (int k = 0; k < H_DIM; ++k) {
      float wv = wlast[(size_t)k * D_OUT + j];
      float sg = 1.0f / (1.0f + __expf(-wv));
      float mm = 2.0f * sg - 1.0f;
      float m2 = mm * mm;
      hb += x4[(size_t)m * H_DIM + k] * mm;
      ds += v4[(size_t)m * H_DIM + k] * m2;
      cs += 1.0f - m2;
    }
    hb += thlast[j];
    ds += cs;
    h = hb * (1.0f / 32.0f) + sqrtf(ds * (1.0f / (float)H_DIM)) * noise[m * D_OUT + j];
    sh[j] = h;
  }
  __syncthreads();
  float mx = sh[0]; int am = 0;
  for (int t = 1; t < D_OUT; ++t) if (sh[t] > mx) { mx = sh[t]; am = t; }
  float se = 0.0f;
  for (int t = 0; t < D_OUT; ++t) se += __expf(sh[t] - mx);
  float lse = mx + __logf(se);
  if (j < D_OUT) {
    out_h [m * D_OUT + j] = h;
    out_lp[m * D_OUT + j] = h - lse;
  }
  if (j == 0) {
    int tg = target[m];
    atomicAdd(out_loss, -(sh[tg] - lse) * (1.0f / (float)M_DIM));
    atomicAdd(out_frac, (am == tg) ? (1.0f / (float)M_DIM) : 0.0f);
  }
}

// ---------------------------------------------------------------------------
// xcov_4[m,i,j] = (i==j) ? v4[m,i] : 0.   536 MB -> 128-bit non-temporal
// streaming stores (exceeds 192MB L2; NT avoids polluting it).
// One f32x4 per thread; r = m*1024 + i == linear index into v4.
// ---------------------------------------------------------------------------
__global__ void fill_xcov_kernel(const float* __restrict__ v4,
                                 float* __restrict__ xcov) {
  size_t t  = (size_t)blockIdx.x * blockDim.x + threadIdx.x;   // < 2^25
  int    j4 = (int)(t & 255);          // which group of 4 columns
  size_t r  = t >> 8;                  // m*1024 + i
  int    i  = (int)(r & 1023);
  int    j  = j4 << 2;
  f32x4  val = {0.0f, 0.0f, 0.0f, 0.0f};
  if (i >= j && i < j + 4) {
    val[i - j] = v4[r];
  }
  __builtin_nontemporal_store(val, (f32x4*)(xcov + (r << 10)) + j4);
}

// ---------------------------------------------------------------------------
extern "C" void kernel_launch(void* const* d_in, const int* in_sizes, int n_in,
                              void* d_out, int out_size, void* d_ws, size_t ws_size,
                              hipStream_t stream) {
  const float* x      = (const float*)d_in[0];
  const int*   target = (const int*)  d_in[1];
  const float* noise  = (const float*)d_in[2];
  const float* w0     = (const float*)d_in[3];
  const float* w1     = (const float*)d_in[4];
  /* w2 = d_in[5] unused (matches reference) */
  const float* w3     = (const float*)d_in[6];
  const float* wlast  = (const float*)d_in[7];
  const float* th0    = (const float*)d_in[8];
  const float* th1    = (const float*)d_in[9];
  /* th2 = d_in[10] unused */
  const float* th3    = (const float*)d_in[11];
  const float* thlast = (const float*)d_in[12];

  // ---- workspace carve-up (all 256B aligned) ----
  char* ws = (char*)d_ws;
  size_t off = 0;
  auto alloc = [&](size_t bytes) -> void* {
    void* p = ws + off;
    off += (bytes + 255) & ~(size_t)255;
    return p;
  };
  _Float16* x16    = (_Float16*)alloc((size_t)M_DIM * K0_PAD * 2);
  _Float16* m0T    = (_Float16*)alloc((size_t)H_DIM * K0_PAD * 2);
  _Float16* m1T    = (_Float16*)alloc((size_t)H_DIM * H_DIM * 2);
  _Float16* m1sqT  = (_Float16*)alloc((size_t)H_DIM * H_DIM * 2);
  _Float16* m3T    = (_Float16*)alloc((size_t)H_DIM * H_DIM * 2);
  _Float16* m3sqT  = (_Float16*)alloc((size_t)H_DIM * H_DIM * 2);
  float*    cs0    = (float*)alloc(H_DIM * 4);
  float*    cs1    = (float*)alloc(H_DIM * 4);
  float*    cs3    = (float*)alloc(H_DIM * 4);
  _Float16* x1f    = (_Float16*)alloc((size_t)M_DIM * H_DIM * 2);
  _Float16* v1f    = (_Float16*)alloc((size_t)M_DIM * H_DIM * 2);
  _Float16* x2f    = (_Float16*)alloc((size_t)M_DIM * H_DIM * 2);
  _Float16* v2f    = (_Float16*)alloc((size_t)M_DIM * H_DIM * 2);
  _Float16* x4f    = (_Float16*)alloc((size_t)M_DIM * H_DIM * 2);
  _Float16* v4f    = (_Float16*)alloc((size_t)M_DIM * H_DIM * 2);
  float*    x4f32  = (float*)alloc((size_t)M_DIM * H_DIM * 4);
  float*    v4f32  = (float*)alloc((size_t)M_DIM * H_DIM * 4);

  // ---- output layout: h(1280) | logprobs(1280) | xcov(128*1024*1024) | loss | frac ----
  float* out      = (float*)d_out;
  float* out_h    = out;
  float* out_lp   = out + (size_t)M_DIM * D_OUT;
  float* out_xcov = out + 2 * (size_t)M_DIM * D_OUT;
  float* out_loss = out_xcov + (size_t)M_DIM * H_DIM * H_DIM;
  float* out_frac = out_loss + 1;

  // 1) input conversion + weight prep
  prep_x_kernel<<<(M_DIM * K0_PAD + 255) / 256, 256, 0, stream>>>(x, x16);
  prep_w_kernel<<<(H_DIM + 255) / 256, 256, 0, stream>>>(w0, m0T, nullptr, cs0, D_IN, H_DIM, K0_PAD);
  prep_w_kernel<<<(H_DIM + 255) / 256, 256, 0, stream>>>(w1, m1T, m1sqT,  cs1, H_DIM, H_DIM, H_DIM);
  prep_w_kernel<<<(H_DIM + 255) / 256, 256, 0, stream>>>(w3, m3T, m3sqT,  cs3, H_DIM, H_DIM, H_DIM);

  // 2) WMMA layers: 8 x (1024/16) = 512 tiles, 8 waves/block -> 64 blocks
  dim3 lgrid(64), lblock(256);
  ebp_layer_wmma<<<lgrid, lblock, 0, stream>>>(x16, nullptr, m0T, nullptr, cs0, th0,
                                               x1f, v1f, nullptr, nullptr, K0_PAD, H_DIM);
  ebp_layer_wmma<<<lgrid, lblock, 0, stream>>>(x1f, v1f, m1T, m1sqT, cs1, th1,
                                               x2f, v2f, nullptr, nullptr, H_DIM, H_DIM);
  ebp_layer_wmma<<<lgrid, lblock, 0, stream>>>(x2f, v2f, m3T, m3sqT, cs3, th3,
                                               x4f, v4f, x4f32, v4f32, H_DIM, H_DIM);

  // 3) last layer + softmax + loss/frac (scalars zeroed first; deterministic)
  zero2_kernel<<<1, 32, 0, stream>>>(out_loss, out_frac);
  last_layer_kernel<<<M_DIM, 32, 0, stream>>>(x4f32, v4f32, wlast, thlast, noise, target,
                                              out_h, out_lp, out_loss, out_frac);

  // 4) the 536 MB xcov stream: 33.55M f32x4 NT stores
  size_t nthreads = ((size_t)M_DIM * H_DIM * H_DIM) / 4;   // 33,554,432
  fill_xcov_kernel<<<(unsigned)(nthreads / 256), 256, 0, stream>>>(v4f32, out_xcov);
}
